// DeepseekV3MLPMoE_78915729096720
// MI455X (gfx1250) — compile-verified, matrix operand order checked by reading, not run on hardware
//
#include <hip/hip_runtime.h>
#include <hip/hip_bf16.h>

// CDNA5 (gfx1250) MoE MLP: fused gate/up + silu*up, then down projection.
// bf16 WMMA (v_wmma_f32_16x16x32_bf16), fp32 accumulate, fp32 in/out.
// LDS rows padded to 40 bf16 (80 B) to break bank conflicts.
// Kernel 2 uses GLOBAL_LOAD_ASYNC_TO_LDS_B128 for the bf16 A tile.

typedef __attribute__((ext_vector_type(16))) __bf16 v16bf;
typedef __attribute__((ext_vector_type(4)))  __bf16 v4bf;
typedef __attribute__((ext_vector_type(8)))  float  v8f;

#define NEXP 32
#define HDIM 2048
#define IDIM 1408
#define TTOK 16384

#define BM 128
#define BN 64
#define BK 32
#define BKP 40        // padded LDS row stride in bf16 (80 B): conflict-free frags
#define THREADS 256   // 8 waves (wave32)

union FragLd { uint4 q[2]; v16bf v; };

__device__ __forceinline__ v16bf load_frag(const __bf16* p) {
    FragLd f;
    f.q[0] = *(const uint4*)p;        // ds_load_b128
    f.q[1] = *(const uint4*)(p + 8);  // ds_load_b128
    return f.v;
}

// ---------------------------------------------------------------------------
// Kernel 1: act[t, n] = silu(hidden @ gate_w[e]) * (hidden @ up_w[e]), bf16 out
// ---------------------------------------------------------------------------
__global__ __launch_bounds__(THREADS)
void moe_gateup_kernel(const float* __restrict__ hidden,
                       const float* __restrict__ gate_w,
                       const float* __restrict__ up_w,
                       const int*   __restrict__ group_sizes,
                       __bf16*      __restrict__ act)
{
    const int e     = blockIdx.z;
    const int mtile = blockIdx.y;
    const int n0    = blockIdx.x * BN;

    // Ragged group offsets: prefix-sum on device (E=32 scalar loads).
    int rowStart = 0;
    for (int i = 0; i < e; ++i) rowStart += group_sizes[i];
    const int rows = group_sizes[e];
    if (mtile * BM >= rows) return;               // uniform early exit
    const int tokenBase = rowStart + mtile * BM;
    const int validRows = min(BM, rows - mtile * BM);

    __shared__ alignas(64) __bf16 Alds[BM * BKP]; // 10 KB, row-major [m][k]
    __shared__ alignas(64) __bf16 Bg[BN * BKP];   // 5 KB, transposed [n][k]
    __shared__ alignas(64) __bf16 Bu[BN * BKP];   // 5 KB, transposed [n][k]

    const int tid    = threadIdx.x;
    const int lane   = tid & 31;
    const int wave   = tid >> 5;
    const int laneLo = lane & 15;
    const int laneHi = lane >> 4;                 // K-half select for A/B frags
    const int fragRow = wave * 16;                // wave owns 16 rows x 64 cols

    v8f accG[4], accU[4];
#pragma unroll
    for (int c = 0; c < 4; ++c) {
        accG[c] = (v8f){0.f,0.f,0.f,0.f,0.f,0.f,0.f,0.f};
        accU[c] = (v8f){0.f,0.f,0.f,0.f,0.f,0.f,0.f,0.f};
    }

    const float* gbase = gate_w + (size_t)e * HDIM * IDIM;
    const float* ubase = up_w   + (size_t)e * HDIM * IDIM;

    for (int k0 = 0; k0 < HDIM; k0 += BK) {
        // ---- stage A: [BM x BK] fp32 -> bf16, branchless (clamped row + 0/1 mask)
#pragma unroll
        for (int j = 0; j < 4; ++j) {
            const int i  = tid + j * THREADS;     // 1024 float4 chunks
            const int r  = i >> 3;
            const int c4 = (i & 7) * 4;
            const int rc = min(r, validRows - 1);
            const float msk = (r < validRows) ? 1.f : 0.f;
            const float4 v = *(const float4*)
                &hidden[(size_t)(tokenBase + rc) * HDIM + (k0 + c4)];
            v4bf p = { (__bf16)(v.x * msk), (__bf16)(v.y * msk),
                       (__bf16)(v.z * msk), (__bf16)(v.w * msk) };
            *(v4bf*)&Alds[r * BKP + c4] = p;      // one 8-byte ds_store
        }
        // ---- stage B (gate & up): coalesced float4 along N, transposed bf16 stores
#pragma unroll
        for (int j = 0; j < 2; ++j) {
            const int i  = tid + j * THREADS;     // 512 float4 chunks
            const int k  = i >> 4;
            const int n4 = (i & 15) * 4;
            const size_t gi = (size_t)(k0 + k) * IDIM + (n0 + n4);
            const float4 g = *(const float4*)&gbase[gi];
            const float4 u = *(const float4*)&ubase[gi];
            Bg[(n4 + 0) * BKP + k] = (__bf16)g.x;
            Bg[(n4 + 1) * BKP + k] = (__bf16)g.y;
            Bg[(n4 + 2) * BKP + k] = (__bf16)g.z;
            Bg[(n4 + 3) * BKP + k] = (__bf16)g.w;
            Bu[(n4 + 0) * BKP + k] = (__bf16)u.x;
            Bu[(n4 + 1) * BKP + k] = (__bf16)u.y;
            Bu[(n4 + 2) * BKP + k] = (__bf16)u.z;
            Bu[(n4 + 3) * BKP + k] = (__bf16)u.w;
        }
        if (k0 + BK < HDIM) {
            __builtin_prefetch(&gbase[(size_t)(k0 + BK) * IDIM + n0], 0, 1);
            __builtin_prefetch(&ubase[(size_t)(k0 + BK) * IDIM + n0], 0, 1);
        }
        __syncthreads();

        // A fragment: row (fragRow+laneLo), 16 contiguous bf16, K-half = laneHi
        const v16bf a = load_frag(&Alds[(fragRow + laneLo) * BKP + laneHi * 16]);
#pragma unroll
        for (int c = 0; c < 4; ++c) {
            const v16bf bg = load_frag(&Bg[(c * 16 + laneLo) * BKP + laneHi * 16]);
            accG[c] = __builtin_amdgcn_wmma_f32_16x16x32_bf16(
                          false, a, false, bg, (short)0, accG[c], false, false);
            const v16bf bu = load_frag(&Bu[(c * 16 + laneLo) * BKP + laneHi * 16]);
            accU[c] = __builtin_amdgcn_wmma_f32_16x16x32_bf16(
                          false, a, false, bu, (short)0, accU[c], false, false);
        }
        __syncthreads();
    }

    // ---- epilogue: silu(gate) * up -> bf16 act ----
#pragma unroll
    for (int c = 0; c < 4; ++c) {
#pragma unroll
        for (int r = 0; r < 8; ++r) {
            const int m = fragRow + laneHi * 8 + r;   // lanes 16-31 => M+8
            if (m < validRows) {
                const float g = accG[c][r];
                const float u = accU[c][r];
                const float s = g / (1.0f + __expf(-g));
                act[(size_t)(tokenBase + m) * IDIM + (n0 + c * 16 + laneLo)] =
                    (__bf16)(s * u);
            }
        }
    }
}

// ---------------------------------------------------------------------------
// Kernel 2: out[t, n] = act @ down_w[e], fp32 out
// ---------------------------------------------------------------------------
__global__ __launch_bounds__(THREADS)
void moe_down_kernel(const __bf16* __restrict__ act,
                     const float*  __restrict__ down_w,
                     const int*    __restrict__ group_sizes,
                     float*        __restrict__ out)
{
    const int e     = blockIdx.z;
    const int mtile = blockIdx.y;
    const int n0    = blockIdx.x * BN;

    int rowStart = 0;
    for (int i = 0; i < e; ++i) rowStart += group_sizes[i];
    const int rows = group_sizes[e];
    if (mtile * BM >= rows) return;
    const int tokenBase = rowStart + mtile * BM;
    const int validRows = min(BM, rows - mtile * BM);

    __shared__ alignas(64) __bf16 Alds[BM * BKP]; // 10 KB
    __shared__ alignas(64) __bf16 Bd[BN * BKP];   // 5 KB, transposed [n][k]

    const int tid    = threadIdx.x;
    const int lane   = tid & 31;
    const int wave   = tid >> 5;
    const int laneLo = lane & 15;
    const int laneHi = lane >> 4;
    const int fragRow = wave * 16;

    v8f acc[4];
#pragma unroll
    for (int c = 0; c < 4; ++c)
        acc[c] = (v8f){0.f,0.f,0.f,0.f,0.f,0.f,0.f,0.f};

    const float* dbase = down_w + (size_t)e * IDIM * HDIM;
    const bool fullTile = (validRows == BM);      // block-uniform

    for (int k0 = 0; k0 < IDIM; k0 += BK) {       // 1408 / 32 = 44 steps
        // ---- stage A: bf16 act tile, 16 B per lane ----
        if (fullTile) {
            // CDNA5 async copy straight into LDS (no VGPR round-trip),
            // tracked by ASYNCcnt.
#pragma unroll
            for (int j = 0; j < 2; ++j) {
                const int i = tid + j * THREADS;  // 512 chunks of 8 bf16
                const int r = i >> 2;
                const int c = (i & 3) * 8;
                const unsigned ldsa =
                    (unsigned)(uintptr_t)&Alds[r * BKP + c];
                const __bf16* ga =
                    &act[(size_t)(tokenBase + r) * IDIM + (k0 + c)];
                asm volatile("global_load_async_to_lds_b128 %0, %1, off"
                             :: "v"(ldsa), "v"(ga) : "memory");
            }
        } else {
            // Ragged edge: branchless clamped copy with zero select.
#pragma unroll
            for (int j = 0; j < 2; ++j) {
                const int i = tid + j * THREADS;
                const int r = i >> 2;
                const int c = (i & 3) * 8;
                const int rc = min(r, validRows - 1);
                uint4 v = *(const uint4*)
                    &act[(size_t)(tokenBase + rc) * IDIM + (k0 + c)];
                if (r >= validRows) v = make_uint4(0u, 0u, 0u, 0u);
                *(uint4*)&Alds[r * BKP + c] = v;
            }
        }
        // ---- stage B: down_w fp32 -> bf16, transposed ----
#pragma unroll
        for (int j = 0; j < 2; ++j) {
            const int i  = tid + j * THREADS;
            const int k  = i >> 4;
            const int n4 = (i & 15) * 4;
            const float4 d = *(const float4*)
                &dbase[(size_t)(k0 + k) * HDIM + (n0 + n4)];
            Bd[(n4 + 0) * BKP + k] = (__bf16)d.x;
            Bd[(n4 + 1) * BKP + k] = (__bf16)d.y;
            Bd[(n4 + 2) * BKP + k] = (__bf16)d.z;
            Bd[(n4 + 3) * BKP + k] = (__bf16)d.w;
        }
        if (k0 + BK < IDIM)
            __builtin_prefetch(&dbase[(size_t)(k0 + BK) * HDIM + n0], 0, 1);

        if (fullTile)
            asm volatile("s_wait_asynccnt 0" ::: "memory");
        __syncthreads();

        const v16bf a = load_frag(&Alds[(fragRow + laneLo) * BKP + laneHi * 16]);
#pragma unroll
        for (int c = 0; c < 4; ++c) {
            const v16bf b = load_frag(&Bd[(c * 16 + laneLo) * BKP + laneHi * 16]);
            acc[c] = __builtin_amdgcn_wmma_f32_16x16x32_bf16(
                         false, a, false, b, (short)0, acc[c], false, false);
        }
        __syncthreads();
    }

#pragma unroll
    for (int c = 0; c < 4; ++c) {
#pragma unroll
        for (int r = 0; r < 8; ++r) {
            const int m = fragRow + laneHi * 8 + r;
            if (m < validRows)
                out[(size_t)(tokenBase + m) * HDIM + (n0 + c * 16 + laneLo)] = acc[c][r];
        }
    }
}

// ---------------------------------------------------------------------------
extern "C" void kernel_launch(void* const* d_in, const int* in_sizes, int n_in,
                              void* d_out, int out_size, void* d_ws, size_t ws_size,
                              hipStream_t stream)
{
    const float* hidden      = (const float*)d_in[0];
    const float* gate_w      = (const float*)d_in[1];
    const float* up_w        = (const float*)d_in[2];
    const float* down_w      = (const float*)d_in[3];
    const int*   group_sizes = (const int*)d_in[4];
    float*       out         = (float*)d_out;
    __bf16*      act         = (__bf16*)d_ws;     // T*I bf16 = 46 MB scratch

    dim3 block(THREADS);
    // Worst case: all tokens on one expert -> T/BM row tiles per expert.
    dim3 g1(IDIM / BN, (TTOK + BM - 1) / BM, NEXP);  // 22 x 128 x 32
    moe_gateup_kernel<<<g1, block, 0, stream>>>(hidden, gate_w, up_w, group_sizes, act);

    dim3 g2(HDIM / BN, (TTOK + BM - 1) / BM, NEXP);  // 32 x 128 x 32
    moe_down_kernel<<<g2, block, 0, stream>>>(act, down_w, group_sizes, out);
}
